// ScaledDotProductAttention_335007449225
// MI455X (gfx1250) — matrix operand order repaired and checked by reading, not compiled, not run on hardware
//
#include <hip/hip_runtime.h>

// CDNA5 / gfx1250 scaled-dot-product attention with explicit score output.
// context = softmax(QK^T/sqrt(128) - 100*mask) V ; also emits the softmax matrix.

typedef __attribute__((ext_vector_type(16))) _Float16 v16h;
typedef __attribute__((ext_vector_type(8)))  float    v8f;

namespace {
constexpr int kB  = 32;
constexpr int kTq = 2048;
constexpr int kTk = 1024;
constexpr int kD  = 128;
constexpr int kDv = 256;
constexpr float kScale = 0.08838834764831845f; // 1/sqrt(128)
}

__global__ __launch_bounds__(256)
void attn_wmma_kernel(const float* __restrict__ Q,
                      const float* __restrict__ K,
                      const float* __restrict__ V,
                      const float* __restrict__ M,
                      float* __restrict__ ctx_out,
                      float* __restrict__ score_out) {
  // 16 query rows x full Tk score panel in LDS (exactly 64KB).
  __shared__ float sS[16 * kTk];

  const int b    = blockIdx.y;
  const int q0   = blockIdx.x * 16;
  const int tid  = threadIdx.x;
  const int wave = tid >> 5;     // 0..7
  const int lane = tid & 31;     // wave32
  const int ln16 = lane & 15;
  const int hi   = lane >> 4;    // 0 or 1

  const float* Qb = Q + ((size_t)b * kTq + q0) * kD;
  const float* Kb = K + (size_t)b * kTk * kD;
  const float* Vb = V + (size_t)b * kTk * kDv;
  const float* Mb = M + ((size_t)b * kTq + q0) * kTk;

  // ---------------- Phase 1: S = QK^T * scale - 100*mask ----------------
  // A-fragments for the 16x128 Q tile: 4 chunks of K=32.
  // Contiguous k order used here (16-bit A 16x32): M = lane&15; K = hi*16 + h.
  v16h afrag[4];
#pragma unroll
  for (int c = 0; c < 4; ++c) {
#pragma unroll
    for (int h = 0; h < 16; ++h) {
      const int kk = c * 32 + hi * 16 + h;
      afrag[c][h] = (_Float16)Qb[ln16 * kD + kk];
    }
  }

  // Each wave handles 8 of the 64 key tiles.
  for (int kt = wave; kt < kTk / 16; kt += 8) {
    v8f acc = {};
#pragma unroll
    for (int c = 0; c < 4; ++c) {
      // B-fragment (32x16 f16): N = lane&15 (key row), K = hi*16 + h (d dim).
      const float* Krow = Kb + (size_t)(kt * 16 + ln16) * kD + c * 32 + hi * 16;
      v16h bfrag;
#pragma unroll
      for (int h = 0; h < 16; ++h) bfrag[h] = (_Float16)Krow[h];
      acc = __builtin_amdgcn_wmma_f32_16x16x32_f16(
          false, afrag[c], false, bfrag, (short)0, acc, false, false);
    }
    // C layout: M = r + hi*8, N = lane&15. Fuse scale + additive mask.
#pragma unroll
    for (int r = 0; r < 8; ++r) {
      const int row = r + hi * 8;
      const int col = kt * 16 + ln16;
      sS[row * kTk + col] = acc[r] * kScale - 100.0f * Mb[row * kTk + col];
    }
  }
  __syncthreads();

  // ---------------- Phase 2: row softmax (each wave owns 2 rows) --------
#pragma unroll
  for (int rr = 0; rr < 2; ++rr) {
    const int row = wave * 2 + rr;
    float* srowLds = sS + row * kTk;

    float m = -3.0e38f;
    for (int c = lane; c < kTk; c += 32) m = fmaxf(m, srowLds[c]);
#pragma unroll
    for (int off = 16; off; off >>= 1) m = fmaxf(m, __shfl_xor(m, off, 32));

    float sum = 0.0f;
    for (int c = lane; c < kTk; c += 32) {
      const float e = __expf(srowLds[c] - m);
      srowLds[c] = e;
      sum += e;
    }
#pragma unroll
    for (int off = 16; off; off >>= 1) sum += __shfl_xor(sum, off, 32);
    const float inv = 1.0f / sum;

    float* srowOut = score_out + ((size_t)b * kTq + q0 + row) * kTk;
    for (int c = lane; c < kTk; c += 32) {
      const float p = srowLds[c] * inv;
      srowLds[c] = p;
      srowOut[c] = p;   // coalesced 128B/wave stores of the softmax output
    }
  }
  __syncthreads();

  // ---------------- Phase 3: context = P V (each wave owns 2 n-tiles) ---
  v8f acc0 = {}, acc1 = {};
  const int n0 = (wave * 2 + 0) * 16 + ln16;
  const int n1 = (wave * 2 + 1) * 16 + ln16;

  for (int step = 0; step < kTk / 32; ++step) {
    const int kbase = step * 32;
    // A from LDS probabilities (f32 -> f16).
    v16h ap;
#pragma unroll
    for (int h = 0; h < 16; ++h) {
      const int kk = kbase + hi * 16 + h;
      ap[h] = (_Float16)sS[ln16 * kTk + kk];
    }
    // B from V (32 k-rows x 16 v-cols), coalesced across lanes per h.
    v16h b0, b1;
#pragma unroll
    for (int h = 0; h < 16; ++h) {
      const size_t vrow = (size_t)(kbase + hi * 16 + h) * kDv;
      b0[h] = (_Float16)Vb[vrow + n0];
      b1[h] = (_Float16)Vb[vrow + n1];
    }
    acc0 = __builtin_amdgcn_wmma_f32_16x16x32_f16(
        false, ap, false, b0, (short)0, acc0, false, false);
    acc1 = __builtin_amdgcn_wmma_f32_16x16x32_f16(
        false, ap, false, b1, (short)0, acc1, false, false);
  }

#pragma unroll
  for (int r = 0; r < 8; ++r) {
    const int row = r + hi * 8;
    float* crow = ctx_out + ((size_t)b * kTq + q0 + row) * kDv;
    crow[n0] = acc0[r];
    crow[n1] = acc1[r];
  }
}

extern "C" void kernel_launch(void* const* d_in, const int* in_sizes, int n_in,
                              void* d_out, int out_size, void* d_ws, size_t ws_size,
                              hipStream_t stream) {
  (void)in_sizes; (void)n_in; (void)out_size; (void)d_ws; (void)ws_size;
  const float* q    = (const float*)d_in[0];
  const float* k    = (const float*)d_in[1];
  const float* v    = (const float*)d_in[2];
  const float* mask = (const float*)d_in[3];

  float* ctx   = (float*)d_out;                          // [32,2048,256]
  float* score = ctx + (size_t)kB * kTq * kDv;           // [32,2048,1024]

  dim3 grid(kTq / 16, kB);   // 128 x 32 = 4096 workgroups
  attn_wmma_kernel<<<grid, 256, 0, stream>>>(q, k, v, mask, ctx, score);
}
